// XPosAttention_65687229825352
// MI455X (gfx1250) — matrix operand order repaired and checked by reading, not compiled, not run on hardware
//
#include <hip/hip_runtime.h>
#include <hip/hip_fp16.h>
#include <math.h>
#include <stdint.h>

typedef __attribute__((ext_vector_type(16))) _Float16 v16h;
typedef __attribute__((ext_vector_type(8)))  _Float16 v8h;
typedef __attribute__((ext_vector_type(8)))  float    v8f;
typedef __attribute__((ext_vector_type(4)))  float    f32x4;
typedef __attribute__((ext_vector_type(4)))  int      i32x4;

#define S_LEN 2048
#define DM    2048
#define NH    16
#define NKV   4
#define HD    128
#define LROW  40   // LDS row pitch in halfs (32 data + 8 pad = 80B, bank-safe)

// ---- async global->LDS staging (graceful fallback to plain stores) --------
#if defined(__has_builtin)
#if __has_builtin(__builtin_amdgcn_global_load_async_to_lds_b128)
#define USE_ASYNC_LDS 1
#endif
#endif

typedef __attribute__((address_space(1))) i32x4 glb_i32x4;
typedef __attribute__((address_space(3))) i32x4 lds_i32x4;

__device__ __forceinline__ void copy16B_to_lds(_Float16* ldst, const _Float16* gsrc)
{
#ifdef USE_ASYNC_LDS
    // generic LDS pointer: low 32 bits are the LDS offset (ISA 10.2 aperture map)
    __builtin_amdgcn_global_load_async_to_lds_b128(
        (glb_i32x4*)(uintptr_t)gsrc,
        (lds_i32x4*)(uint32_t)(uintptr_t)ldst, 0, 0);
#else
    *(v8h*)ldst = *(const v8h*)gsrc;
#endif
}

template <int N>
__device__ __forceinline__ void async_wait()
{
#ifdef USE_ASYNC_LDS
#if __has_builtin(__builtin_amdgcn_s_wait_asynccnt)
    __builtin_amdgcn_s_wait_asynccnt(N);
#else
    asm volatile("s_wait_asynccnt %0" :: "n"(N) : "memory");
#endif
#endif
}

__device__ __forceinline__ v16h join8(v8h a, v8h b)
{
    return __builtin_shufflevector(a, b, 0,1,2,3,4,5,6,7,8,9,10,11,12,13,14,15);
}

// ---------------------------------------------------------------------------
// C[M,N](f32) = A[M,K](f16) @ Bt[N,K](f16)^T.  Block: 8 waves, 128(M)x64(N).
// Double-buffered LDS tiles filled with ASYNCcnt-tracked copies: tile i+1
// streams in (3 async b128 per wave) while tile i feeds 4 WMMAs per wave.
// ---------------------------------------------------------------------------
__global__ __launch_bounds__(256)
void gemm_f16_wmma(const _Float16* __restrict__ A, const _Float16* __restrict__ Bt,
                   float* __restrict__ C, int M, int N, int K)
{
    __shared__ _Float16 lA[2][128 * LROW];
    __shared__ _Float16 lB[2][64 * LROW];

    const int tid  = threadIdx.x;
    const int wave = tid >> 5;
    const int lane = tid & 31;
    const int l16  = lane & 15;
    const int half = lane >> 4;
    const int n0   = blockIdx.x * 64;
    const int m0   = blockIdx.y * 128;
    const int mw   = m0 + wave * 16;

    const int row = tid >> 2, ch = tid & 3;        // cooperative-load coords
    auto issue_tile = [&](int buf, int k0) {
        // A tile: 128 rows x 32 halfs (2 chunks/thread), B tile: 64 rows (1 chunk)
        copy16B_to_lds(&lA[buf][row * LROW + ch * 8],
                       A + (size_t)(m0 + row) * K + k0 + ch * 8);
        copy16B_to_lds(&lA[buf][(row + 64) * LROW + ch * 8],
                       A + (size_t)(m0 + row + 64) * K + k0 + ch * 8);
        copy16B_to_lds(&lB[buf][row * LROW + ch * 8],
                       Bt + (size_t)(n0 + row) * K + k0 + ch * 8);
    };

    v8f acc[4];
    #pragma unroll
    for (int t = 0; t < 4; ++t)
        #pragma unroll
        for (int r = 0; r < 8; ++r) acc[t][r] = 0.0f;

    issue_tile(0, 0);                              // prologue: tile 0 in flight

    int cur = 0;
    for (int k0 = 0; k0 < K; k0 += 32) {
        if (k0 + 32 < K) {
            issue_tile(cur ^ 1, k0 + 32);          // overlap next tile w/ compute
            __builtin_prefetch(A  + (size_t)(m0 + row) * K + k0 + 64, 0, 0);
            __builtin_prefetch(Bt + (size_t)(n0 + row) * K + k0 + 64, 0, 0);
            async_wait<3>();                       // in-order: tile k0 resident
        } else {
            async_wait<0>();
        }
        __syncthreads();                           // whole block sees tile k0

        // A frag: row = wave*16+l16; elems 0..7 @ half*8, 8..15 @ half*8+16
        const _Float16* ar = &lA[cur][(wave * 16 + l16) * LROW + half * 8];
        v16h a = join8(*(const v8h*)ar, *(const v8h*)(ar + 16));

        #pragma unroll
        for (int t = 0; t < 4; ++t) {
            // B frag: col = t*16+l16; elems 0..7 @ half*8, 8..15 @ half*8+16
            const _Float16* br = &lB[cur][(t * 16 + l16) * LROW + half * 8];
            v16h b = join8(*(const v8h*)br, *(const v8h*)(br + 16));
            acc[t] = __builtin_amdgcn_wmma_f32_16x16x32_f16(false, a, false, b,
                                                            (short)0, acc[t], false, false);
        }
        __syncthreads();                           // reads done before buf reuse
        cur ^= 1;
    }

    #pragma unroll
    for (int t = 0; t < 4; ++t) {
        float* Crow = C + (size_t)(mw + half * 8) * N + (n0 + t * 16 + l16);
        #pragma unroll
        for (int r = 0; r < 8; ++r) Crow[(size_t)r * N] = acc[t][r];
    }
}

// ---------------------------------------------------------------------------
// prep kernels: x -> f16; W(K,N) f32 -> Wt(N,K) f16 (LDS-tiled transpose)
// ---------------------------------------------------------------------------
__global__ __launch_bounds__(256)
void convert_f16_kernel(const float* __restrict__ in, _Float16* __restrict__ out)
{
    size_t i = ((size_t)blockIdx.x * blockDim.x + threadIdx.x) * 4;
    f32x4 v = *(const f32x4*)(in + i);
    out[i + 0] = (_Float16)v[0]; out[i + 1] = (_Float16)v[1];
    out[i + 2] = (_Float16)v[2]; out[i + 3] = (_Float16)v[3];
}

__global__ __launch_bounds__(256)
void transpose_f16_kernel(const float* __restrict__ in, _Float16* __restrict__ out,
                          int K, int N)
{
    __shared__ float tile[32][33];
    int kb = blockIdx.y * 32, nb = blockIdx.x * 32;
    int tx = threadIdx.x, ty = threadIdx.y;          // (32, 8)
    #pragma unroll
    for (int i = 0; i < 32; i += 8)
        tile[ty + i][tx] = in[(size_t)(kb + ty + i) * N + nb + tx];
    __syncthreads();
    #pragma unroll
    for (int i = 0; i < 32; i += 8)
        out[(size_t)(nb + ty + i) * K + kb + tx] = (_Float16)tile[tx][ty + i];
}

// ---------------------------------------------------------------------------
// RoPE + XPos, f32 (B,S,H,D) -> f16 (B,H,S,D).  Q additionally folds 1/sqrt(HD).
// ---------------------------------------------------------------------------
__global__ __launch_bounds__(256)
void rope_xpos_kernel(const float* __restrict__ in, _Float16* __restrict__ out,
                      int nheads, int is_q)
{
    size_t idx = (size_t)blockIdx.x * blockDim.x + threadIdx.x;
    int d = (int)(idx & (HD - 1));
    size_t t = idx >> 7;
    int h = (int)(t % nheads); t /= nheads;
    int s = (int)(t % S_LEN);
    int b = (int)(t / S_LEN);

    const float* src = in + ((size_t)(b * S_LEN + s) * nheads + h) * HD;
    float x0 = src[d];
    int dh = d & 63;
    float ang = (float)s * expf(-(float)dh * (9.210340371976184f / 64.0f));
    float c = cosf(ang), sn = sinf(ang);
    float other = (d < 64) ? -src[d + 64] : src[d - 64];
    float r = x0 * c + other * sn;

    float sc = ((float)s + 256.0f) / 512.0f;
    float xs = ((d & 1) == 0) ? sc : 1.0f / sc;
    r *= is_q ? xs : 1.0f / xs;
    if (is_q) r *= 0.08838834764831845f;             // 1/sqrt(128)

    out[((size_t)(b * nheads + h) * S_LEN + s) * HD + d] = (_Float16)r;
}

// V: f32 (B,S,NKV,HD) -> f16 (B,NKV,HD,S)  (key-contiguous for PV B-frags)
__global__ __launch_bounds__(256)
void v_transpose_kernel(const float* __restrict__ in, _Float16* __restrict__ out)
{
    size_t idx = (size_t)blockIdx.x * blockDim.x + threadIdx.x;
    int s = (int)(idx % S_LEN);
    size_t t = idx / S_LEN;
    int d  = (int)(t % HD); t /= HD;
    int hk = (int)(t % NKV);
    int b  = (int)(t / NKV);
    out[idx] = (_Float16)in[((size_t)(b * S_LEN + s) * NKV + hk) * HD + d];
}

// ---------------------------------------------------------------------------
// Flash attention, all-f16 operands, f32 accum.  One wave per (b,h,16 queries).
//  Q:(B,NH,S,HD) K:(B,NKV,S,HD) Vt:(B,NKV,HD,S) -> O f16 (B,S,NH*HD)
// ---------------------------------------------------------------------------
__global__ __launch_bounds__(32)
void flash_attn_kernel(const _Float16* __restrict__ Q, const _Float16* __restrict__ K,
                       const _Float16* __restrict__ Vt, _Float16* __restrict__ O)
{
    __shared__ _Float16 lds_p[16 * 32];

    const int lane = threadIdx.x;
    const int l16  = lane & 15;
    const int half = lane >> 4;
    const int q0   = blockIdx.x * 16;
    const int h    = blockIdx.y;
    const int b    = blockIdx.z;
    const int hk   = h >> 2;

    const _Float16* Qrow = Q + ((size_t)(b * NH + h) * S_LEN + q0 + l16) * HD;
    v16h qa[4];
    #pragma unroll
    for (int c = 0; c < 4; ++c) {
        const _Float16* p = Qrow + c * 32 + half * 8;
        qa[c] = join8(*(const v8h*)p, *(const v8h*)(p + 16));
    }

    v8f o[8];
    #pragma unroll
    for (int t = 0; t < 8; ++t)
        #pragma unroll
        for (int r = 0; r < 8; ++r) o[t][r] = 0.0f;
    float rm[8], rl[8];
    #pragma unroll
    for (int r = 0; r < 8; ++r) { rm[r] = -__builtin_inff(); rl[r] = 0.0f; }

    const _Float16* Kbase = K + (size_t)(b * NKV + hk) * S_LEN * HD;
    const int kmax = q0 + 16;
    for (int k0 = 0; k0 < kmax; k0 += 32) {
        if (k0 + 32 < kmax) {                      // cover next 32x256B K tile
            __builtin_prefetch(Kbase + (size_t)(k0 + 32 + l16) * HD + half * 64, 0, 0);
            __builtin_prefetch(Kbase + (size_t)(k0 + 48 + l16) * HD + half * 64, 0, 0);
        }

        // scores: two 16x16 frags over 32 keys
        v8f sc[2];
        #pragma unroll
        for (int t = 0; t < 2; ++t) {
            #pragma unroll
            for (int r = 0; r < 8; ++r) sc[t][r] = 0.0f;
            const _Float16* Krow = Kbase + (size_t)(k0 + t * 16 + l16) * HD;
            #pragma unroll
            for (int c = 0; c < 4; ++c) {
                const _Float16* p = Krow + c * 32 + half * 8;
                v16h kb = join8(*(const v8h*)p, *(const v8h*)(p + 16));
                sc[t] = __builtin_amdgcn_wmma_f32_16x16x32_f16(false, qa[c], false, kb,
                                                               (short)0, sc[t], false, false);
            }
        }

        // causal mask + online softmax (rows r+half*8, 16-lane reductions)
        #pragma unroll
        for (int r = 0; r < 8; ++r) {
            const int row = q0 + r + half * 8;
            float v0 = (k0 + l16      <= row) ? sc[0][r] : -__builtin_inff();
            float v1 = (k0 + 16 + l16 <= row) ? sc[1][r] : -__builtin_inff();
            float mx = fmaxf(v0, v1);
            #pragma unroll
            for (int d = 8; d >= 1; d >>= 1) mx = fmaxf(mx, __shfl_xor(mx, d, 32));
            float mnew  = fmaxf(rm[r], mx);
            float alpha = __expf(rm[r] - mnew);
            float p0 = __expf(v0 - mnew);
            float p1 = __expf(v1 - mnew);
            float rs = p0 + p1;
            #pragma unroll
            for (int d = 8; d >= 1; d >>= 1) rs += __shfl_xor(rs, d, 32);
            rl[r] = rl[r] * alpha + rs;
            rm[r] = mnew;
            #pragma unroll
            for (int t = 0; t < 8; ++t) o[t][r] *= alpha;
            lds_p[(r + half * 8) * 32 + l16]      = (_Float16)p0;
            lds_p[(r + half * 8) * 32 + 16 + l16] = (_Float16)p1;
        }

        // C-frag -> A-frag re-layout through LDS (single wave, DS in-order)
        const _Float16* prow = &lds_p[l16 * 32 + half * 8];
        v16h pa = join8(*(const v8h*)prow, *(const v8h*)(prow + 16));

        // O += P(16x32) . V(32x128); Vt rows are key-contiguous
        #pragma unroll
        for (int t = 0; t < 8; ++t) {
            const _Float16* vr =
                Vt + ((size_t)(b * NKV + hk) * HD + t * 16 + l16) * S_LEN + k0 + half * 8;
            v16h vb = join8(*(const v8h*)vr, *(const v8h*)(vr + 16));
            o[t] = __builtin_amdgcn_wmma_f32_16x16x32_f16(false, pa, false, vb,
                                                          (short)0, o[t], false, false);
        }
    }

    #pragma unroll
    for (int r = 0; r < 8; ++r) {
        float inv = 1.0f / rl[r];
        const int row = q0 + r + half * 8;
        _Float16* orow = O + (size_t)(b * S_LEN + row) * (NH * HD) + h * HD + l16;
        #pragma unroll
        for (int t = 0; t < 8; ++t) orow[t * 16] = (_Float16)(o[t][r] * inv);
    }
}

// ---------------------------------------------------------------------------
extern "C" void kernel_launch(void* const* d_in, const int* in_sizes, int n_in,
                              void* d_out, int out_size, void* d_ws, size_t ws_size,
                              hipStream_t stream)
{
    const float* x  = (const float*)d_in[0];
    const float* Wq = (const float*)d_in[1];
    const float* Wk = (const float*)d_in[2];
    const float* Wv = (const float*)d_in[3];
    const float* Wo = (const float*)d_in[4];
    float* out = (float*)d_out;

    char* ws = (char*)d_ws;
    const size_t MB = (size_t)1 << 20;
    _Float16* xh     = (_Float16*)(ws);             // 16 MB (M,K) f16
    _Float16* Wqt    = (_Float16*)(ws + 16  * MB);  //  8 MB (2048,2048)
    _Float16* Wkt    = (_Float16*)(ws + 24  * MB);  //  2 MB (512,2048)
    _Float16* Wvt    = (_Float16*)(ws + 26  * MB);  //  2 MB
    _Float16* Wot    = (_Float16*)(ws + 28  * MB);  //  8 MB
    float*    q_proj = (float*)   (ws + 36  * MB);  // 32 MB f32
    float*    k_proj = (float*)   (ws + 68  * MB);  //  8 MB
    float*    v_proj = (float*)   (ws + 76  * MB);  //  8 MB
    _Float16* qh     = (_Float16*)(ws + 84  * MB);  // 16 MB (B,NH,S,HD)
    _Float16* kh     = (_Float16*)(ws + 100 * MB);  //  4 MB (B,NKV,S,HD)
    _Float16* vth    = (_Float16*)(ws + 104 * MB);  //  4 MB (B,NKV,HD,S)
    _Float16* attn_o = (_Float16*)q_proj;           // reuse (16 MB needed)

    const int M = 2 * S_LEN;                         // 4096 token rows

    // --- precision/layout prep ---
    convert_f16_kernel<<<(M * DM / 4) / 256, 256, 0, stream>>>(x, xh);
    dim3 tb(32, 8);
    transpose_f16_kernel<<<dim3(DM / 32, DM / 32), tb, 0, stream>>>(Wq, Wqt, DM, DM);
    transpose_f16_kernel<<<dim3((NKV * HD) / 32, DM / 32), tb, 0, stream>>>(Wk, Wkt, DM, NKV * HD);
    transpose_f16_kernel<<<dim3((NKV * HD) / 32, DM / 32), tb, 0, stream>>>(Wv, Wvt, DM, NKV * HD);
    transpose_f16_kernel<<<dim3(DM / 32, DM / 32), tb, 0, stream>>>(Wo, Wot, DM, DM);

    // --- projections ---
    gemm_f16_wmma<<<dim3((NH * HD) / 64, M / 128), 256, 0, stream>>>(xh, Wqt, q_proj, M, NH * HD, DM);
    gemm_f16_wmma<<<dim3((NKV * HD) / 64, M / 128), 256, 0, stream>>>(xh, Wkt, k_proj, M, NKV * HD, DM);
    gemm_f16_wmma<<<dim3((NKV * HD) / 64, M / 128), 256, 0, stream>>>(xh, Wvt, v_proj, M, NKV * HD, DM);

    // --- RoPE/XPos + layout for attention ---
    rope_xpos_kernel<<<(2 * S_LEN * NH * HD) / 256, 256, 0, stream>>>(q_proj, qh, NH, 1);
    rope_xpos_kernel<<<(2 * S_LEN * NKV * HD) / 256, 256, 0, stream>>>(k_proj, kh, NKV, 0);
    v_transpose_kernel<<<(2 * S_LEN * NKV * HD) / 256, 256, 0, stream>>>(v_proj, vth);

    // --- attention ---
    flash_attn_kernel<<<dim3(S_LEN / 16, NH, 2), 32, 0, stream>>>(qh, kh, vth, attn_o);

    // --- output projection ---
    gemm_f16_wmma<<<dim3(DM / 64, M / 128), 256, 0, stream>>>(attn_o, Wot, out, M, DM, DM);
}
